// Attention_69518340653293
// MI455X (gfx1250) — matrix operand order repaired and checked by reading, not compiled, not run on hardware
//
#include <hip/hip_runtime.h>
#include <hip/hip_bf16.h>

typedef unsigned short u16;
typedef __attribute__((ext_vector_type(16))) __bf16 v16bf;
typedef __attribute__((ext_vector_type(8)))  float  v8f;

union FragU { v16bf v; uint4 q[2]; };

__device__ inline u16 f2bf(float f) {
  unsigned u = __builtin_bit_cast(unsigned, f);
  unsigned r = u + 0x7FFFu + ((u >> 16) & 1u);   // round-to-nearest-even
  return (u16)(r >> 16);
}

__device__ inline v8f wmma_bf16(v16bf a, v16bf b, v8f c) {
  return __builtin_amdgcn_wmma_f32_16x16x32_bf16(false, a, false, b, (short)0, c,
                                                 false, false);
}

// A fragment (16x32, MxK): lane = M row; K elements at [hi*8 .. +7] and [16+hi*8 .. +7]
__device__ inline v16bf load_fragA(const u16* base, int ld, int row0, int k0) {
  unsigned lane = threadIdx.x & 31u;
  unsigned hi = lane >> 4;
  const u16* p = base + (size_t)(row0 + (lane & 15u)) * ld + k0 + hi * 8u;
  FragU f;
  f.q[0] = *(const uint4*)p;
  f.q[1] = *(const uint4*)(p + 16);
  return f.v;
}

// B fragment (32x16, KxN) gathered from an N-major (row=N, K contiguous) matrix:
// lane = N col; K elements at [hi*8 .. +7] and [16+hi*8 .. +7]
__device__ inline v16bf load_fragB(const u16* base, int ld, int col0, int k0) {
  unsigned lane = threadIdx.x & 31u;
  unsigned hi = lane >> 4;
  const u16* p = base + (size_t)(col0 + (lane & 15u)) * ld + k0 + hi * 8u;
  FragU f;
  f.q[0] = *(const uint4*)p;
  f.q[1] = *(const uint4*)(p + 16);
  return f.v;
}

// ---------------- conversion kernels ----------------
__global__ __launch_bounds__(256) void conv_bf16(const float* __restrict__ in,
                                                 u16* __restrict__ out, int n) {
  int i = (blockIdx.x * blockDim.x + threadIdx.x) * 4;
  if (i < n) {
    float4 v = *(const float4*)(in + i);
    u16 o0 = f2bf(v.x), o1 = f2bf(v.y), o2 = f2bf(v.z), o3 = f2bf(v.w);
    ushort4 o; o.x = o0; o.y = o1; o.z = o2; o.w = o3;
    *(ushort4*)(out + i) = o;
  }
}

// in: R x C fp32 row-major  ->  out: C x R bf16 row-major (i.e. transposed)
__global__ __launch_bounds__(256) void transpose_conv(const float* __restrict__ in,
                                                      u16* __restrict__ out,
                                                      int R, int C) {
  int idx = blockIdx.x * blockDim.x + threadIdx.x;
  int r = idx % R, c = idx / R;
  if (c < C) out[(size_t)c * R + r] = f2bf(in[(size_t)r * C + c]);
}

// ---------------- QKV projection GEMM ----------------
// Xb: 4096x1024 bf16, Wt: 3072x1024 bf16 (N-major). Each wave: 64x64 tile.
// Scatter epilogue: Q,K as [bh][s][64]; V transposed as [bh][64][s] (all bf16).
__global__ __launch_bounds__(256) void qkv_gemm(const u16* __restrict__ Xb,
                                                const u16* __restrict__ Wt,
                                                const float* __restrict__ bqkv,
                                                u16* __restrict__ Q,
                                                u16* __restrict__ K,
                                                u16* __restrict__ Vt) {
  const int NT = 3072 / 64;  // 48 column tiles
  unsigned wave = blockIdx.x * (blockDim.x >> 5) + (threadIdx.x >> 5);
  int row0 = (int)(wave / NT) * 64;
  int col0 = (int)(wave % NT) * 64;
  v8f acc[4][4] = {};
  for (int k0 = 0; k0 < 1024; k0 += 32) {
    v16bf a[4], b[4];
#pragma unroll
    for (int i = 0; i < 4; i++) a[i] = load_fragA(Xb, 1024, row0 + 16 * i, k0);
#pragma unroll
    for (int j = 0; j < 4; j++) b[j] = load_fragB(Wt, 1024, col0 + 16 * j, k0);
#pragma unroll
    for (int i = 0; i < 4; i++)
#pragma unroll
      for (int j = 0; j < 4; j++) acc[i][j] = wmma_bf16(a[i], b[j], acc[i][j]);
  }
  unsigned lane = threadIdx.x & 31u;
  unsigned hi = lane >> 4, ln = lane & 15u;
#pragma unroll
  for (int j = 0; j < 4; j++) {
    int n = col0 + 16 * j + (int)ln;
    float bias = bqkv[n];
#pragma unroll
    for (int i = 0; i < 4; i++) {
#pragma unroll
      for (int r = 0; r < 8; r++) {
        int row = row0 + 16 * i + 8 * (int)hi + r;        // global M = b*2048+s
        int bb = row >> 11, s = row & 2047;
        u16 bv = f2bf(acc[i][j][r] + bias);
        if (n < 1024) {
          int h = n >> 6, d = n & 63;
          Q[(((size_t)(bb * 16 + h)) * 2048 + s) * 64 + d] = bv;
        } else if (n < 2048) {
          int n2 = n - 1024, h = n2 >> 6, d = n2 & 63;
          K[(((size_t)(bb * 16 + h)) * 2048 + s) * 64 + d] = bv;
        } else {
          int n2 = n - 2048, h = n2 >> 6, d = n2 & 63;
          Vt[(((size_t)(bb * 16 + h)) * 64 + d) * 2048 + s] = bv;
        }
      }
    }
  }
}

// ---------------- flash attention ----------------
// One wave per 32 query rows of one (b,h). Streams keys in tiles of 32.
// K/V fragments are loaded once per key tile and reused for both 16-row
// M-subtiles -> 16 WMMAs per 16 b128 loads (32 FLOP/byte vs L2).
__global__ __launch_bounds__(256) void flash_attn(const u16* __restrict__ Q,
                                                  const u16* __restrict__ K,
                                                  const u16* __restrict__ Vt,
                                                  u16* __restrict__ Ob) {
  __shared__ __align__(16) u16 pstage[8][32 * 32];  // 2KB per wave
  unsigned wid = threadIdx.x >> 5;
  unsigned lane = threadIdx.x & 31u;
  unsigned hi = lane >> 4, ln = lane & 15u;
  unsigned gw = blockIdx.x * 8u + wid;
  unsigned bh = gw >> 6;                 // 0..31 (64 qtiles per bh)
  int q0 = (int)(gw & 63u) * 32;         // query tile start (32 rows)
  const u16* Qb = Q + (size_t)bh * 2048 * 64;
  const u16* Kb = K + (size_t)bh * 2048 * 64;
  const u16* Vb = Vt + (size_t)bh * 64 * 2048;

  v16bf qf[2][2];  // [mi][kk]
#pragma unroll
  for (int mi = 0; mi < 2; mi++) {
    qf[mi][0] = load_fragA(Qb, 64, q0 + 16 * mi, 0);
    qf[mi][1] = load_fragA(Qb, 64, q0 + 16 * mi, 32);
  }

  v8f o[2][4] = {};          // [mi][d-tile]
  float mrun[2][8], lrun[2][8];
#pragma unroll
  for (int mi = 0; mi < 2; mi++)
#pragma unroll
    for (int r = 0; r < 8; r++) { mrun[mi][r] = -1e30f; lrun[mi][r] = 0.f; }
  const float scale = 0.125f;  // 1/sqrt(64)
  u16* ps = &pstage[wid][0];

  for (int t0 = 0; t0 < 2048; t0 += 32) {
    // K fragments: loaded once, shared by both M-subtiles
    v16bf kf[2][2];  // [nj][kk]
#pragma unroll
    for (int nj = 0; nj < 2; nj++)
#pragma unroll
      for (int kk = 0; kk < 2; kk++)
        kf[nj][kk] = load_fragB(Kb, 64, t0 + 16 * nj, 32 * kk);

    v8f sc[2][2] = {};  // [mi][nj]
#pragma unroll
    for (int mi = 0; mi < 2; mi++)
#pragma unroll
      for (int nj = 0; nj < 2; nj++)
#pragma unroll
        for (int kk = 0; kk < 2; kk++)
          sc[mi][nj] = wmma_bf16(qf[mi][kk], kf[nj][kk], sc[mi][nj]);

    // online softmax per M-subtile
#pragma unroll
    for (int mi = 0; mi < 2; mi++) {
      float corr[8];
#pragma unroll
      for (int r = 0; r < 8; r++) {
        float s0 = sc[mi][0][r] * scale, s1 = sc[mi][1][r] * scale;
        float mx = fmaxf(s0, s1);
        mx = fmaxf(mx, __shfl_xor(mx, 1, 32));
        mx = fmaxf(mx, __shfl_xor(mx, 2, 32));
        mx = fmaxf(mx, __shfl_xor(mx, 4, 32));
        mx = fmaxf(mx, __shfl_xor(mx, 8, 32));
        float mnew = fmaxf(mrun[mi][r], mx);
        float c = __expf(mrun[mi][r] - mnew);
        float p0 = __expf(s0 - mnew);
        float p1 = __expf(s1 - mnew);
        float rs = p0 + p1;
        rs += __shfl_xor(rs, 1, 32);
        rs += __shfl_xor(rs, 2, 32);
        rs += __shfl_xor(rs, 4, 32);
        rs += __shfl_xor(rs, 8, 32);
        lrun[mi][r] = lrun[mi][r] * c + rs;
        mrun[mi][r] = mnew;
        corr[r] = c;
        unsigned rowi = 16u * (unsigned)mi + 8u * hi + (unsigned)r;
        ps[rowi * 32 + ln] = f2bf(p0);
        ps[rowi * 32 + 16 + ln] = f2bf(p1);
      }
#pragma unroll
      for (int j = 0; j < 4; j++)
#pragma unroll
        for (int r = 0; r < 8; r++) o[mi][j][r] *= corr[r];
    }

    // V fragments: loaded once, shared by both M-subtiles
    v16bf vf[4];
#pragma unroll
    for (int j = 0; j < 4; j++) vf[j] = load_fragB(Vb, 2048, 16 * j, t0);

    // restage P (32x32) from C-layout to A-layout via LDS, then PV
#pragma unroll
    for (int mi = 0; mi < 2; mi++) {
      FragU pf;
      const u16* pp = ps + (16u * mi + ln) * 32 + hi * 8;
      pf.q[0] = *(const uint4*)pp;
      pf.q[1] = *(const uint4*)(pp + 16);
#pragma unroll
      for (int j = 0; j < 4; j++)
        o[mi][j] = wmma_bf16(pf.v, vf[j], o[mi][j]);
    }
  }

  unsigned b = bh >> 4, h = bh & 15u;
#pragma unroll
  for (int mi = 0; mi < 2; mi++)
#pragma unroll
    for (int j = 0; j < 4; j++)
#pragma unroll
      for (int r = 0; r < 8; r++) {
        float v = o[mi][j][r] / lrun[mi][r];
        size_t row = (size_t)b * 2048 + (size_t)(q0 + 16 * mi + 8 * (int)hi + r);
        Ob[row * 1024 + h * 64 + 16 * j + ln] = f2bf(v);
      }
}

// ---------------- output projection GEMM ----------------
// Ob: 4096x1024 bf16, Wot: 1024x1024 bf16 (N-major). fp32 out + bias.
__global__ __launch_bounds__(256) void out_gemm(const u16* __restrict__ Ob,
                                                const u16* __restrict__ Wot,
                                                const float* __restrict__ bout,
                                                float* __restrict__ out) {
  const int NT = 1024 / 64;  // 16
  unsigned wave = blockIdx.x * (blockDim.x >> 5) + (threadIdx.x >> 5);
  int row0 = (int)(wave / NT) * 64;
  int col0 = (int)(wave % NT) * 64;
  v8f acc[4][4] = {};
  for (int k0 = 0; k0 < 1024; k0 += 32) {
    v16bf a[4], b[4];
#pragma unroll
    for (int i = 0; i < 4; i++) a[i] = load_fragA(Ob, 1024, row0 + 16 * i, k0);
#pragma unroll
    for (int j = 0; j < 4; j++) b[j] = load_fragB(Wot, 1024, col0 + 16 * j, k0);
#pragma unroll
    for (int i = 0; i < 4; i++)
#pragma unroll
      for (int j = 0; j < 4; j++) acc[i][j] = wmma_bf16(a[i], b[j], acc[i][j]);
  }
  unsigned lane = threadIdx.x & 31u;
  unsigned hi = lane >> 4, ln = lane & 15u;
#pragma unroll
  for (int j = 0; j < 4; j++) {
    int n = col0 + 16 * j + (int)ln;
    float bias = bout[n];
#pragma unroll
    for (int i = 0; i < 4; i++)
#pragma unroll
      for (int r = 0; r < 8; r++) {
        int row = row0 + 16 * i + 8 * (int)hi + r;
        out[(size_t)row * 1024 + n] = acc[i][j][r] + bias;
      }
  }
}

extern "C" void kernel_launch(void* const* d_in, const int* in_sizes, int n_in,
                              void* d_out, int out_size, void* d_ws, size_t ws_size,
                              hipStream_t stream) {
  (void)in_sizes; (void)n_in; (void)out_size; (void)ws_size;
  const float* x     = (const float*)d_in[0];  // (2,2048,1024)
  const float* W_qkv = (const float*)d_in[1];  // (1024,3072)
  const float* b_qkv = (const float*)d_in[2];  // (3072,)
  const float* W_out = (const float*)d_in[3];  // (1024,1024)
  const float* b_out = (const float*)d_in[4];  // (1024,)
  float* out = (float*)d_out;

  const size_t MB = 1u << 20;
  char* ws = (char*)d_ws;
  u16* Xb  = (u16*)(ws + 0 * MB);    // 4096x1024 bf16  (8 MiB)
  u16* Wqt = (u16*)(ws + 8 * MB);    // 3072x1024 bf16  (6 MiB)
  u16* Wot = (u16*)(ws + 14 * MB);   // 1024x1024 bf16  (2 MiB)
  u16* Qw  = (u16*)(ws + 16 * MB);   // [32][2048][64]  (8 MiB)
  u16* Kw  = (u16*)(ws + 24 * MB);   // [32][2048][64]  (8 MiB)
  u16* Vtw = (u16*)(ws + 32 * MB);   // [32][64][2048]  (8 MiB)
  u16* Obw = (u16*)(ws + 40 * MB);   // 4096x1024 bf16  (8 MiB)  => 48 MiB total

  conv_bf16<<<4096 * 1024 / (256 * 4), 256, 0, stream>>>(x, Xb, 4096 * 1024);
  transpose_conv<<<(3072 * 1024) / 256, 256, 0, stream>>>(W_qkv, Wqt, 1024, 3072);
  transpose_conv<<<(1024 * 1024) / 256, 256, 0, stream>>>(W_out, Wot, 1024, 1024);

  // 64x48 = 3072 wave tiles, 8 waves per block
  qkv_gemm<<<384, 256, 0, stream>>>(Xb, Wqt, b_qkv, Qw, Kw, Vtw);

  // 32 bh * 64 qtiles = 2048 waves, 8 per block
  flash_attn<<<256, 256, 0, stream>>>(Qw, Kw, Vtw, Obw);

  // 64x16 = 1024 wave tiles, 8 per block
  out_gemm<<<128, 256, 0, stream>>>(Obw, Wot, b_out, out);
}
